// SemanticFarthestPointSampling_21165598835476
// MI455X (gfx1250) — matrix operand order repaired and checked by reading, not compiled, not run on hardware
//
#include <hip/hip_runtime.h>
#include <hip/hip_bf16.h>

typedef _Float16 h16;
typedef __attribute__((ext_vector_type(16))) _Float16 v16h;
typedef __attribute__((ext_vector_type(8)))  _Float16 v8h;
typedef __attribute__((ext_vector_type(8)))  float    v8f;
typedef __attribute__((ext_vector_type(4)))  float    v4f;

#define B_   4
#define N_   16384
#define C_   256
#define H_   256
#define M_   4096
#define EPS_ 1e-5f
#define INF_ 1e10f

#if __has_builtin(__builtin_amdgcn_global_load_async_to_lds_b32) && \
    __has_builtin(__builtin_amdgcn_s_wait_asynccnt)
#define HAVE_ASYNC_LDS 1
#endif

// ---------------------------------------------------------------------------
// prep: fold BN stats, transpose W1/W2 into f16 N-major layout for WMMA B-frags
// ---------------------------------------------------------------------------
__global__ void prep_kernel(const float* __restrict__ W1, const float* __restrict__ W2,
                            const float* __restrict__ s1, const float* __restrict__ bb1,
                            const float* __restrict__ m1, const float* __restrict__ v1,
                            const float* __restrict__ s2, const float* __restrict__ bb2,
                            const float* __restrict__ m2, const float* __restrict__ v2,
                            h16* __restrict__ Wt1, h16* __restrict__ Wt2,
                            float* __restrict__ a1, float* __restrict__ c1,
                            float* __restrict__ a2, float* __restrict__ c2)
{
    int t = blockIdx.x * 256 + threadIdx.x;   // 0 .. 65535
    int n = t >> 8;
    int k = t & 255;
    Wt1[t] = (h16)W1[k * H_ + n];             // Wt[n][k] = W[k][n]
    Wt2[t] = (h16)W2[k * H_ + n];
    if (t < H_) {
        float A1 = s1[t] * rsqrtf(v1[t] + EPS_);
        a1[t] = A1; c1[t] = bb1[t] - m1[t] * A1;
        float A2 = s2[t] * rsqrtf(v2[t] + EPS_);
        a2[t] = A2; c2[t] = bb2[t] - m2[t] * A2;
    }
}

// ---------------------------------------------------------------------------
// mlp: 64 rows/block, 8 waves. WMMA f32_16x16x32_f16 for both GEMM stages.
// wave w: row strip (w>>1), column half (w&1) (8 col tiles of 16).
// A fragments register-resident; B fragments 2-deep double-buffered;
// two interleaved accumulator chains to fill the WMMA hazard window.
// ---------------------------------------------------------------------------
__launch_bounds__(256, 1)
__global__ void mlp_kernel(const float* __restrict__ feats,
                           const h16* __restrict__ Wt1,
                           const h16* __restrict__ Wt2,
                           const float* __restrict__ a1, const float* __restrict__ c1,
                           const float* __restrict__ a2, const float* __restrict__ c2,
                           const float* __restrict__ W3, const float* __restrict__ b3,
                           float* __restrict__ scores, float* __restrict__ weights)
{
    __shared__ h16   act[64 * H_];     // stage-1 output tile (f16), 32 KB
    __shared__ float sscore[64];
    __shared__ float s_a1[H_], s_c1[H_], s_a2[H_], s_c2[H_], s_w3[H_];

    const int tid   = threadIdx.x;
    const int wave  = tid >> 5;
    const int lane  = tid & 31;
    const int strip = wave >> 1;       // 0..3, 16 rows each
    const int half  = wave & 1;        // col tiles half*8 .. half*8+7
    const int lrow  = lane & 15;
    const int kg    = lane >> 4;

    const long long row0 = (long long)blockIdx.x * 64;

    if (tid < 64) sscore[tid] = 0.f;

    // stage epilogue constants into LDS (async path when toolchain supports it)
#ifdef HAVE_ASYNC_LDS
    if (tid < H_) {
        __builtin_amdgcn_global_load_async_to_lds_b32((int*)(a1 + tid), (int*)&s_a1[tid], 0, 0);
        __builtin_amdgcn_global_load_async_to_lds_b32((int*)(c1 + tid), (int*)&s_c1[tid], 0, 0);
        __builtin_amdgcn_global_load_async_to_lds_b32((int*)(a2 + tid), (int*)&s_a2[tid], 0, 0);
        __builtin_amdgcn_global_load_async_to_lds_b32((int*)(c2 + tid), (int*)&s_c2[tid], 0, 0);
        __builtin_amdgcn_global_load_async_to_lds_b32((int*)(W3 + tid), (int*)&s_w3[tid], 0, 0);
    }
    __builtin_amdgcn_s_wait_asynccnt(0);
#else
    if (tid < H_) {
        s_a1[tid] = a1[tid]; s_c1[tid] = c1[tid];
        s_a2[tid] = a2[tid]; s_c2[tid] = c2[tid];
        s_w3[tid] = W3[tid];
    }
#endif
    __syncthreads();

    const float* arow = feats + (row0 + strip * 16 + lrow) * (long long)C_;

    // ------- Stage 1: preload all 8 A fragments (f32 -> f16) -------------
    v16h afrag[8];
    #pragma unroll
    for (int k = 0; k < 8; ++k) {
        const float* p0 = arow + k * 32 + kg * 16;
        v4f f0 = *(const v4f*)(p0);
        v4f f1 = *(const v4f*)(p0 + 4);
        v4f f2 = *(const v4f*)(p0 + 8);
        v4f f3 = *(const v4f*)(p0 + 12);
        v16h a;
        #pragma unroll
        for (int i = 0; i < 4; ++i) {
            a[i]      = (h16)f0[i];
            a[4 + i]  = (h16)f1[i];
            a[8 + i]  = (h16)f2[i];
            a[12 + i] = (h16)f3[i];
        }
        afrag[k] = a;
    }

    // ------- Stage 1: feats @ W1 -> BN1 -> ReLU -> LDS (f16) -------------
    for (int j = 0; j < 8; ++j) {
        const int t = half * 8 + j;
        const h16* bbase = Wt1 + (t * 16 + lrow) * H_ + kg * 16;
        // warm stage-2 B tile for this t while stage 1 computes
        __builtin_prefetch(Wt2 + (t * 16 + lrow) * H_ + kg * 16, 0, 3);

        v16h b0 = *(const v16h*)(bbase);
        v16h b1 = *(const v16h*)(bbase + 32);
        v8f acc0 = {}, acc1 = {};
        #pragma unroll
        for (int k = 0; k < 8; k += 2) {
            v16h n0 = (k < 6) ? *(const v16h*)(bbase + (k + 2) * 32) : b0;
            v16h n1 = (k < 6) ? *(const v16h*)(bbase + (k + 3) * 32) : b1;
            acc0 = __builtin_amdgcn_wmma_f32_16x16x32_f16(
                       false, afrag[k],     false, b0, (short)0, acc0, false, false);
            acc1 = __builtin_amdgcn_wmma_f32_16x16x32_f16(
                       false, afrag[k + 1], false, b1, (short)0, acc1, false, false);
            b0 = n0; b1 = n1;
        }
        // epilogue: merge chains, BN1 + ReLU, store f16 activation tile
        const int n  = t * 16 + lrow;
        const float sa = s_a1[n], sc = s_c1[n];
        #pragma unroll
        for (int jj = 0; jj < 8; ++jj) {
            int   m = strip * 16 + jj + kg * 8;
            float h = (acc0[jj] + acc1[jj]) * sa + sc;
            h = h > 0.f ? h : 0.f;
            act[m * H_ + n] = (h16)h;
        }
    }
    __syncthreads();

    // ------- Stage 2: preload A fragments from LDS activation tile -------
    #pragma unroll
    for (int k = 0; k < 8; ++k) {
        const h16* p0 = act + (strip * 16 + lrow) * H_ + k * 32 + kg * 16;
        v8h lo = *(const v8h*)(p0);
        v8h hi = *(const v8h*)(p0 + 8);
        v16h a;
        #pragma unroll
        for (int i = 0; i < 8; ++i) { a[i] = lo[i]; a[8 + i] = hi[i]; }
        afrag[k] = a;
    }

    // ------- Stage 2: act @ W2 -> BN2 -> ReLU -> dot W3 ------------------
    float part[8];
    #pragma unroll
    for (int i = 0; i < 8; ++i) part[i] = 0.f;

    for (int j = 0; j < 8; ++j) {
        const int t = half * 8 + j;
        const h16* bbase = Wt2 + (t * 16 + lrow) * H_ + kg * 16;

        v16h b0 = *(const v16h*)(bbase);
        v16h b1 = *(const v16h*)(bbase + 32);
        v8f acc0 = {}, acc1 = {};
        #pragma unroll
        for (int k = 0; k < 8; k += 2) {
            v16h n0 = (k < 6) ? *(const v16h*)(bbase + (k + 2) * 32) : b0;
            v16h n1 = (k < 6) ? *(const v16h*)(bbase + (k + 3) * 32) : b1;
            acc0 = __builtin_amdgcn_wmma_f32_16x16x32_f16(
                       false, afrag[k],     false, b0, (short)0, acc0, false, false);
            acc1 = __builtin_amdgcn_wmma_f32_16x16x32_f16(
                       false, afrag[k + 1], false, b1, (short)0, acc1, false, false);
            b0 = n0; b1 = n1;
        }
        const int n  = t * 16 + lrow;
        const float sa = s_a2[n], sc = s_c2[n], w3 = s_w3[n];
        #pragma unroll
        for (int jj = 0; jj < 8; ++jj) {
            float h = (acc0[jj] + acc1[jj]) * sa + sc;
            h = h > 0.f ? h : 0.f;
            part[jj] += h * w3;
        }
    }

    // reduce over the 16 columns held by each half-wave (masks 1,2,4,8 stay in group)
    #pragma unroll
    for (int off = 1; off < 16; off <<= 1) {
        #pragma unroll
        for (int jj = 0; jj < 8; ++jj)
            part[jj] += __shfl_xor(part[jj], off, 32);
    }
    if (lrow == 0) {
        #pragma unroll
        for (int jj = 0; jj < 8; ++jj)
            atomicAdd(&sscore[strip * 16 + jj + kg * 8], part[jj]);
    }
    __syncthreads();

    if (tid < 64) {
        float s = sscore[tid] + b3[0];
        long long r = row0 + tid;
        scores[r]  = s;
        weights[r] = 1.f / (1.f + __expf(-s));
    }
}

// ---------------------------------------------------------------------------
// fps: one persistent block per batch. xyz/w/temp fully register-resident.
// ---------------------------------------------------------------------------
__launch_bounds__(1024, 1)
__global__ void fps_kernel(const float* __restrict__ xyz,
                           const float* __restrict__ weights,
                           long long* __restrict__ idx_out)
{
    __shared__ float redv[32];
    __shared__ int   redi[32];
    __shared__ float lastp[3];

    const int b    = blockIdx.x;
    const int tid  = threadIdx.x;
    const int lane = tid & 31;
    const int wid  = tid >> 5;

    const float* px = xyz     + (size_t)b * N_ * 3;
    const float* wb = weights + (size_t)b * N_;

    float x[16], y[16], z[16], w[16], temp[16];
    #pragma unroll
    for (int i = 0; i < 16; ++i) {
        int p = tid + i * 1024;
        x[i] = px[p * 3 + 0];
        y[i] = px[p * 3 + 1];
        z[i] = px[p * 3 + 2];
        w[i] = wb[p];
        temp[i] = INF_;
    }
    if (tid == 0) {
        idx_out[(size_t)b * M_] = 0;
        lastp[0] = px[0]; lastp[1] = px[1]; lastp[2] = px[2];
    }
    __syncthreads();

    for (int m = 1; m < M_; ++m) {
        const float lx = lastp[0], ly = lastp[1], lz = lastp[2];
        float bv = -1.f;
        int   bi = 0;
        #pragma unroll
        for (int i = 0; i < 16; ++i) {
            float dx = x[i] - lx, dy = y[i] - ly, dz = z[i] - lz;
            float d2 = (dx * dx + dy * dy + dz * dz) * w[i];
            temp[i] = fminf(temp[i], d2);
            if (temp[i] > bv) { bv = temp[i]; bi = tid + i * 1024; }
        }
        // wave32 argmax (first-index tie break)
        #pragma unroll
        for (int off = 1; off < 32; off <<= 1) {
            float ov = __shfl_xor(bv, off, 32);
            int   oi = __shfl_xor(bi, off, 32);
            if (ov > bv || (ov == bv && oi < bi)) { bv = ov; bi = oi; }
        }
        if (lane == 0) { redv[wid] = bv; redi[wid] = bi; }
        __syncthreads();
        if (wid == 0) {
            bv = redv[lane];
            bi = redi[lane];
            #pragma unroll
            for (int off = 1; off < 32; off <<= 1) {
                float ov = __shfl_xor(bv, off, 32);
                int   oi = __shfl_xor(bi, off, 32);
                if (ov > bv || (ov == bv && oi < bi)) { bv = ov; bi = oi; }
            }
            if (lane == 0) {
                idx_out[(size_t)b * M_ + m] = (long long)bi;
                lastp[0] = px[(size_t)bi * 3 + 0];
                lastp[1] = px[(size_t)bi * 3 + 1];
                lastp[2] = px[(size_t)bi * 3 + 2];
            }
        }
        __syncthreads();
    }
}

// ---------------------------------------------------------------------------
extern "C" void kernel_launch(void* const* d_in, const int* in_sizes, int n_in,
                              void* d_out, int out_size, void* d_ws, size_t ws_size,
                              hipStream_t stream)
{
    const float* xyz   = (const float*)d_in[0];
    const float* feats = (const float*)d_in[1];
    const float* W1    = (const float*)d_in[2];
    const float* s1    = (const float*)d_in[3];
    const float* bb1   = (const float*)d_in[4];
    const float* m1    = (const float*)d_in[5];
    const float* v1    = (const float*)d_in[6];
    const float* W2    = (const float*)d_in[7];
    const float* s2    = (const float*)d_in[8];
    const float* bb2   = (const float*)d_in[9];
    const float* m2    = (const float*)d_in[10];
    const float* v2    = (const float*)d_in[11];
    const float* W3    = (const float*)d_in[12];
    const float* b3    = (const float*)d_in[13];

    // workspace layout
    char* ws = (char*)d_ws;
    h16*   Wt1     = (h16*)(ws);                       // 128 KB
    h16*   Wt2     = (h16*)(ws + 131072);              // 128 KB
    float* a1      = (float*)(ws + 262144);            // 4 x 1 KB fold arrays
    float* c1      = a1 + H_;
    float* a2      = c1 + H_;
    float* c2      = a2 + H_;
    float* wgt     = (float*)(ws + 262144 + 4096);     // B*N f32 sigmoid weights

    // output layout: [B*M int64 idx][B*N f32 scores]
    long long* idx_out = (long long*)d_out;
    float*     scores  = (float*)((char*)d_out + sizeof(long long) * (size_t)B_ * M_);

    prep_kernel<<<(H_ * H_) / 256, 256, 0, stream>>>(
        W1, W2, s1, bb1, m1, v1, s2, bb2, m2, v2, Wt1, Wt2, a1, c1, a2, c2);

    mlp_kernel<<<(B_ * N_) / 64, 256, 0, stream>>>(
        feats, Wt1, Wt2, a1, c1, a2, c2, W3, b3, scores, wgt);

    fps_kernel<<<B_, 1024, 0, stream>>>(xyz, wgt, idx_out);
}
